// LRU_84842783965444
// MI455X (gfx1250) — matrix operand order repaired and checked by reading, not compiled, not run on hardware
//
#include <hip/hip_runtime.h>

// LRU forward: Bu GEMMs (bf16 WMMA) -> chunked complex scan -> output GEMM (bf16 WMMA) + D*x
#define LL   8192
#define HH   1024
#define NNC  1024
#define NCH  64
#define CLEN 128   // NCH * CLEN == LL

typedef __attribute__((ext_vector_type(16))) __bf16 v16bf;
typedef __attribute__((ext_vector_type(8)))  __bf16 v8bf;
typedef __attribute__((ext_vector_type(8)))  float  v8f;

static __device__ __forceinline__ v8bf ld8(const __bf16* p) { return *(const v8bf*)p; }
static __device__ __forceinline__ v16bf cat16(v8bf lo, v8bf hi) {
  return __builtin_shufflevector(lo, hi, 0,1,2,3,4,5,6,7,8,9,10,11,12,13,14,15);
}

// A fragment: 16x32 bf16 tile of row-major A (lda), rows row0..row0+15, K cols k0..k0+31.
// Lane l: M = l&15, half = l>>4; elements 0..7 = K (half*8 .. half*8+7),
// elements 8..15 = K (16+half*8 .. 16+half*8+7)   [ISA 05_wmma 7.12.2 16-bit A layout]
static __device__ __forceinline__ v16bf a_frag(const __bf16* A, int lda, int row0, int k0, int lane) {
  int m = lane & 15, half = lane >> 4;
  const __bf16* p = A + (size_t)(row0 + m) * lda + (k0 + half * 8);
  return cat16(ld8(p), ld8(p + 16));
}
// B fragment: 32x16 (KxN); element (k, n) = M[(n0+n)*ldb + k].
// Lane l: N = l&15, half = l>>4; elements 0..7 = K (half*8 .. half*8+7), elements 8..15 = K (16+half*8 .. 16+half*8+7).
static __device__ __forceinline__ v16bf b_frag(const __bf16* M, int ldb, int n0, int k0, int lane) {
  int n = lane & 15, kh = (lane >> 4) << 3;
  const __bf16* p = M + (size_t)(n0 + n) * ldb + (k0 + kh);
  return cat16(ld8(p), ld8(p + 16));
}
static __device__ __forceinline__ v8f wmma_bf16(v16bf a, v16bf b, v8f c) {
  return __builtin_amdgcn_wmma_f32_16x16x32_bf16(false, a, false, b, (short)0, c, false, false);
}

// ---------------- precision / parameter prep ----------------
__global__ void k_cvt_bf16(const float* __restrict__ in, __bf16* __restrict__ out, int n, float scale) {
  int i = blockIdx.x * blockDim.x + threadIdx.x;
  int stride = gridDim.x * blockDim.x;
  for (; i < n; i += stride) out[i] = (__bf16)(in[i] * scale);
}

__global__ void k_lambda(const float* __restrict__ nu_log, const float* __restrict__ theta_log,
                         const float* __restrict__ gamma_log,
                         float* __restrict__ lam_re, float* __restrict__ lam_im, float* __restrict__ gam) {
  int n = blockIdx.x * blockDim.x + threadIdx.x;
  if (n < NNC) {
    float r  = __expf(-__expf(nu_log[n]));
    float th = __expf(theta_log[n]);
    lam_re[n] = r * __cosf(th);
    lam_im[n] = r * __sinf(th);
    gam[n]    = __expf(gamma_log[n]);
  }
}

// ---------------- Bu = (x @ B^T) * gamma, bf16 WMMA, fp32 out ----------------
// grid (LL/32, 4), block 256 (8 waves). Wave tile: 32 rows x 64 cols of 2048 (re|im).
__global__ void __launch_bounds__(256) k_gemm_bu(const __bf16* __restrict__ xbf,
                                                 const __bf16* __restrict__ Bre,
                                                 const __bf16* __restrict__ Bim,
                                                 const float* __restrict__ gam,
                                                 float* __restrict__ BuRe, float* __restrict__ BuIm) {
  int lane = threadIdx.x & 31, wave = threadIdx.x >> 5;
  int row0 = blockIdx.x << 5;
  int col0 = (blockIdx.y * 8 + wave) << 6;      // [0, 2048)
  bool isIm = col0 >= NNC;
  int n0 = isIm ? (col0 - NNC) : col0;
  const __bf16* Bm = isIm ? Bim : Bre;
  float* Out = isIm ? BuIm : BuRe;

  v8f acc[2][4] = {};
  for (int k0 = 0; k0 < HH; k0 += 32) {
    v16bf a0 = a_frag(xbf, HH, row0, k0, lane);
    v16bf a1 = a_frag(xbf, HH, row0 + 16, k0, lane);
#pragma unroll
    for (int s = 0; s < 4; ++s) {
      v16bf b = b_frag(Bm, HH, n0 + s * 16, k0, lane);
      acc[0][s] = wmma_bf16(a0, b, acc[0][s]);
      acc[1][s] = wmma_bf16(a1, b, acc[1][s]);
    }
  }
  int half = lane >> 4, nn = lane & 15;
#pragma unroll
  for (int s = 0; s < 4; ++s) {
    int col = n0 + s * 16 + nn;
    float g = gam[col];
#pragma unroll
    for (int i = 0; i < 2; ++i) {
#pragma unroll
      for (int r = 0; r < 8; ++r) {
        int row = row0 + i * 16 + r + half * 8;
        Out[(size_t)row * NNC + col] = acc[i][s][r] * g;
      }
    }
  }
}

// ---------------- scan phase 1: per-chunk reduction ----------------
__global__ void k_scan_reduce(const float* __restrict__ BuRe, const float* __restrict__ BuIm,
                              const float* __restrict__ lam_re, const float* __restrict__ lam_im,
                              float* __restrict__ cbr, float* __restrict__ cbi) {
  int n = blockIdx.x * blockDim.x + threadIdx.x;   // channel
  int c = blockIdx.y;                              // chunk
  float lr = lam_re[n], li = lam_im[n];
  float hr = 0.f, hi = 0.f;
  size_t base = (size_t)c * CLEN * NNC + n;
  for (int t = 0; t < CLEN; ++t) {
    float br = BuRe[base], bi = BuIm[base];
    float nr = fmaf(lr, hr, fmaf(-li, hi, br));
    float ni = fmaf(lr, hi, fmaf(li, hr, bi));
    hr = nr; hi = ni;
    base += NNC;
  }
  cbr[(size_t)c * NNC + n] = hr;
  cbi[(size_t)c * NNC + n] = hi;
}

// ---------------- scan phase 2: carry scan across chunks (lam^128 by squaring) ----------------
__global__ void k_scan_carry(const float* __restrict__ lam_re, const float* __restrict__ lam_im,
                             const float* __restrict__ cbr, const float* __restrict__ cbi,
                             float* __restrict__ hir, float* __restrict__ hii) {
  int n = blockIdx.x * blockDim.x + threadIdx.x;
  float pr = lam_re[n], pi = lam_im[n];
#pragma unroll
  for (int i = 0; i < 7; ++i) { float a = pr * pr - pi * pi; pi = 2.f * pr * pi; pr = a; } // lam^128
  float hr = 0.f, hi = 0.f;
  for (int c = 0; c < NCH; ++c) {
    size_t o = (size_t)c * NNC + n;
    hir[o] = hr; hii[o] = hi;                       // state entering chunk c
    float nr = fmaf(pr, hr, fmaf(-pi, hi, cbr[o]));
    float ni = fmaf(pr, hi, fmaf(pi, hr, cbi[o]));
    hr = nr; hi = ni;
  }
}

// ---------------- scan phase 3: emit states (bf16 for output GEMM) ----------------
__global__ void k_scan_states(const float* __restrict__ BuRe, const float* __restrict__ BuIm,
                              const float* __restrict__ lam_re, const float* __restrict__ lam_im,
                              const float* __restrict__ hir, const float* __restrict__ hii,
                              __bf16* __restrict__ sre, __bf16* __restrict__ sim) {
  int n = blockIdx.x * blockDim.x + threadIdx.x;
  int c = blockIdx.y;
  float lr = lam_re[n], li = lam_im[n];
  size_t co = (size_t)c * NNC + n;
  float hr = hir[co], hi = hii[co];
  size_t base = (size_t)c * CLEN * NNC + n;
  for (int t = 0; t < CLEN; ++t) {
    float br = BuRe[base], bi = BuIm[base];
    float nr = fmaf(lr, hr, fmaf(-li, hi, br));
    float ni = fmaf(lr, hi, fmaf(li, hr, bi));
    hr = nr; hi = ni;
    sre[base] = (__bf16)hr;
    sim[base] = (__bf16)hi;
    base += NNC;
  }
}

// ---------------- y = s_re @ C_re^T + s_im @ (-C_im)^T + D*x ----------------
// grid (LL/32, HH/512), block 256. Wave tile: 32 rows x 64 cols.
__global__ void __launch_bounds__(256) k_gemm_y(const __bf16* __restrict__ sre,
                                                const __bf16* __restrict__ sim,
                                                const __bf16* __restrict__ Cre,
                                                const __bf16* __restrict__ nCim,
                                                const float* __restrict__ x,
                                                const float* __restrict__ Dv,
                                                float* __restrict__ y) {
  int lane = threadIdx.x & 31, wave = threadIdx.x >> 5;
  int row0 = blockIdx.x << 5;
  int col0 = (blockIdx.y * 8 + wave) << 6;        // [0, 1024)

  v8f acc[2][4] = {};
  for (int k0 = 0; k0 < NNC; k0 += 32) {
    v16bf ar0 = a_frag(sre, NNC, row0, k0, lane);
    v16bf ar1 = a_frag(sre, NNC, row0 + 16, k0, lane);
#pragma unroll
    for (int s = 0; s < 4; ++s) {
      v16bf b = b_frag(Cre, NNC, col0 + s * 16, k0, lane);
      acc[0][s] = wmma_bf16(ar0, b, acc[0][s]);
      acc[1][s] = wmma_bf16(ar1, b, acc[1][s]);
    }
    v16bf ai0 = a_frag(sim, NNC, row0, k0, lane);
    v16bf ai1 = a_frag(sim, NNC, row0 + 16, k0, lane);
#pragma unroll
    for (int s = 0; s < 4; ++s) {
      v16bf b = b_frag(nCim, NNC, col0 + s * 16, k0, lane);
      acc[0][s] = wmma_bf16(ai0, b, acc[0][s]);
      acc[1][s] = wmma_bf16(ai1, b, acc[1][s]);
    }
  }
  int half = lane >> 4, nn = lane & 15;
#pragma unroll
  for (int s = 0; s < 4; ++s) {
    int col = col0 + s * 16 + nn;
    float d = Dv[col];
#pragma unroll
    for (int i = 0; i < 2; ++i) {
#pragma unroll
      for (int r = 0; r < 8; ++r) {
        int row = row0 + i * 16 + r + half * 8;
        size_t o = (size_t)row * HH + col;
        y[o] = acc[i][s][r] + d * x[o];
      }
    }
  }
}

extern "C" void kernel_launch(void* const* d_in, const int* in_sizes, int n_in,
                              void* d_out, int out_size, void* d_ws, size_t ws_size,
                              hipStream_t stream) {
  const float* x         = (const float*)d_in[0];
  const float* nu_log    = (const float*)d_in[1];
  const float* theta_log = (const float*)d_in[2];
  const float* B_re      = (const float*)d_in[3];
  const float* B_im      = (const float*)d_in[4];
  const float* C_re      = (const float*)d_in[5];
  const float* C_im      = (const float*)d_in[6];
  const float* Dv        = (const float*)d_in[7];
  const float* gamma_log = (const float*)d_in[8];
  float* y = (float*)d_out;

  char* ws = (char*)d_ws;
  size_t off = 0;
  auto wsalloc = [&](size_t bytes) -> char* {
    char* p = ws + off;
    off = (off + bytes + 255) & ~(size_t)255;
    return p;
  };
  __bf16* xbf  = (__bf16*)wsalloc((size_t)LL * HH * 2);
  __bf16* Bre  = (__bf16*)wsalloc((size_t)NNC * HH * 2);
  __bf16* Bim  = (__bf16*)wsalloc((size_t)NNC * HH * 2);
  __bf16* Cre  = (__bf16*)wsalloc((size_t)HH * NNC * 2);
  __bf16* nCim = (__bf16*)wsalloc((size_t)HH * NNC * 2);
  float*  lamr = (float*)wsalloc((size_t)NNC * 4);
  float*  lami = (float*)wsalloc((size_t)NNC * 4);
  float*  gam  = (float*)wsalloc((size_t)NNC * 4);
  float*  BuRe = (float*)wsalloc((size_t)LL * NNC * 4);
  float*  BuIm = (float*)wsalloc((size_t)LL * NNC * 4);
  float*  cbr  = (float*)wsalloc((size_t)NCH * NNC * 4);
  float*  cbi  = (float*)wsalloc((size_t)NCH * NNC * 4);
  float*  hir  = (float*)wsalloc((size_t)NCH * NNC * 4);
  float*  hii  = (float*)wsalloc((size_t)NCH * NNC * 4);
  __bf16* sre  = (__bf16*)wsalloc((size_t)LL * NNC * 2);
  __bf16* sim  = (__bf16*)wsalloc((size_t)LL * NNC * 2);
  (void)ws_size; (void)in_sizes; (void)n_in; (void)out_size;

  // precision prep
  k_cvt_bf16<<<2048, 256, 0, stream>>>(x, xbf, LL * HH, 1.0f);
  k_cvt_bf16<<<512, 256, 0, stream>>>(B_re, Bre, NNC * HH, 1.0f);
  k_cvt_bf16<<<512, 256, 0, stream>>>(B_im, Bim, NNC * HH, 1.0f);
  k_cvt_bf16<<<512, 256, 0, stream>>>(C_re, Cre, HH * NNC, 1.0f);
  k_cvt_bf16<<<512, 256, 0, stream>>>(C_im, nCim, HH * NNC, -1.0f); // pre-negate: bf16 WMMA has no B-neg
  k_lambda<<<NNC / 256, 256, 0, stream>>>(nu_log, theta_log, gamma_log, lamr, lami, gam);

  // input GEMMs
  k_gemm_bu<<<dim3(LL / 32, 4), 256, 0, stream>>>(xbf, Bre, Bim, gam, BuRe, BuIm);

  // chunked associative scan
  k_scan_reduce<<<dim3(NNC / 256, NCH), 256, 0, stream>>>(BuRe, BuIm, lamr, lami, cbr, cbi);
  k_scan_carry<<<NNC / 256, 256, 0, stream>>>(lamr, lami, cbr, cbi, hir, hii);
  k_scan_states<<<dim3(NNC / 256, NCH), 256, 0, stream>>>(BuRe, BuIm, lamr, lami, hir, hii, sre, sim);

  // output GEMM + skip connection
  k_gemm_y<<<dim3(LL / 32, HH / 512), 256, 0, stream>>>(sre, sim, Cre, nCim, x, Dv, y);
}
